// crysVQVAE_14474039788285
// MI455X (gfx1250) — matrix operand order repaired and checked by reading, not compiled, not run on hardware
//
#include <hip/hip_runtime.h>
#include <hip/hip_bf16.h>

#define N_NODES  131072
#define N_GRAPHS 2048
#define HIDDEN   256
#define N_CODES  512
#define MAXZ     100
#define CHUNK    32                    // codes per LDS buffer (32*256*2B*2arrays = 32KB/buf)
#define NCHUNKS  (N_CODES / CHUNK)     // 16

typedef __attribute__((ext_vector_type(16))) __bf16 v16bf;
typedef __attribute__((ext_vector_type(8)))  float  v8f;
typedef __attribute__((ext_vector_type(4)))  int    v4i;

union Frag {
    v16bf v;
    uint4 q[2];
    __bf16 h[16];
};

static __device__ __forceinline__ v8f wmma_bf16(const v16bf& a, const v16bf& b, const v8f& c) {
    return __builtin_amdgcn_wmma_f32_16x16x32_bf16(false, a, false, b, (short)0, c, false, false);
}

// ---- CDNA5 async global->LDS copy (ASYNCcnt-tracked), with sync fallback ----
#if defined(__has_builtin)
#if __has_builtin(__builtin_amdgcn_global_load_async_to_lds_b128)
#define HAVE_ASYNC_LDS 1
#endif
#endif
#ifndef HAVE_ASYNC_LDS
#define HAVE_ASYNC_LDS 0
#endif

typedef __attribute__((address_space(1))) v4i* as1_v4i_p;   // printed as "int4 __device__ *"
typedef __attribute__((address_space(3))) v4i* as3_v4i_p;   // printed as "int4 __shared__ *"

static __device__ __forceinline__ void async_copy_b128(const uint4* g, uint4* l) {
#if HAVE_ASYNC_LDS
    uint4* gnc = const_cast<uint4*>(g);
    __builtin_amdgcn_global_load_async_to_lds_b128((as1_v4i_p)gnc, (as3_v4i_p)l, 0, 0);
#else
    *l = *g;
#endif
}

static __device__ __forceinline__ void wait_async_copies() {
#if HAVE_ASYNC_LDS
#if defined(__has_builtin) && __has_builtin(__builtin_amdgcn_s_wait_asynccnt)
    __builtin_amdgcn_s_wait_asynccnt(0);
#else
    asm volatile("s_wait_asynccnt 0" ::: "memory");
#endif
#endif
}

// ---------------------------------------------------------------------------
// Kernel 1: per-code precompute.  One block per code (256 threads).
//   cb_hi/cb_lo : bf16 split of codebook rows
//   cnorm[j]    : ||c_j||^2
//   logits[j,k] : codebook[j] @ W_atom + b_atom      (512 x 100)
//   lse[j]      : logsumexp_k logits[j,k]
// ---------------------------------------------------------------------------
__global__ __launch_bounds__(256) void precompute_codes(
    const float* __restrict__ cb, const float* __restrict__ W_atom,
    const float* __restrict__ b_atom,
    __bf16* __restrict__ cb_hi, __bf16* __restrict__ cb_lo,
    float* __restrict__ cnorm, float* __restrict__ logits, float* __restrict__ lse)
{
    __shared__ float sc[HIDDEN];
    __shared__ float red[256];
    __shared__ float lbuf[MAXZ];
    const int j = blockIdx.x;
    const int t = threadIdx.x;

    float c = cb[j * HIDDEN + t];
    __bf16 h = (__bf16)c;
    cb_hi[j * HIDDEN + t] = h;
    cb_lo[j * HIDDEN + t] = (__bf16)(c - (float)h);
    sc[t]  = c;
    red[t] = c * c;
    __syncthreads();
    for (int s = 128; s > 0; s >>= 1) {
        if (t < s) red[t] += red[t + s];
        __syncthreads();
    }
    if (t == 0) cnorm[j] = red[0];

    if (t < MAXZ) {
        float acc = b_atom[t];
        for (int k = 0; k < HIDDEN; ++k) acc += sc[k] * W_atom[k * MAXZ + t];
        lbuf[t] = acc;
        logits[j * MAXZ + t] = acc;
    }
    __syncthreads();
    if (t == 0) {
        float m = -3.4e38f;
        for (int k = 0; k < MAXZ; ++k) m = lbuf[k] > m ? lbuf[k] : m;
        float s = 0.f;
        for (int k = 0; k < MAXZ; ++k) s += expf(lbuf[k] - m);
        lse[j] = m + logf(s);
    }
}

// ---------------------------------------------------------------------------
// Kernel 2: argmin over codes via bf16x2-split WMMA with double-buffered
// async LDS staging. Block = 8 waves; each wave owns 16 nodes (all 512 codes).
// argmin_j ( ||c_j||^2 - 2 * z . c_j )   (||z||^2 constant per row -> dropped)
// ---------------------------------------------------------------------------
__global__ __launch_bounds__(256) void argmin_kernel(
    const float* __restrict__ zt, const __bf16* __restrict__ cb_hi,
    const __bf16* __restrict__ cb_lo, const float* __restrict__ cnorm,
    int* __restrict__ idx_out)
{
    __shared__ __bf16 lds_hi[2][CHUNK * HIDDEN];   // 2 x 16KB
    __shared__ __bf16 lds_lo[2][CHUNK * HIDDEN];   // 2 x 16KB

    const int tid  = threadIdx.x;
    const int lane = tid & 31;
    const int wid  = tid >> 5;
    const int nodeBase = (blockIdx.x * 8 + wid) * 16;
    const int mrow = nodeBase + (lane & 15);

    // A fragments (16 nodes x 256 K), split into hi/lo bf16. Resident in VGPRs.
    Frag Ah[8], Al[8];
    {
        const float* zrow = zt + (size_t)mrow * HIDDEN;
        const int koff = (lane >= 16) ? 8 : 0;
        #pragma unroll
        for (int kt = 0; kt < 8; ++kt) {
            const int klo = kt * 32 + koff;
            #pragma unroll
            for (int i = 0; i < 8; ++i) {
                float f0 = zrow[klo + i];
                float f1 = zrow[klo + 16 + i];
                __bf16 h0 = (__bf16)f0, h1 = (__bf16)f1;
                Ah[kt].h[i]     = h0;
                Ah[kt].h[8 + i] = h1;
                Al[kt].h[i]     = (__bf16)(f0 - (float)h0);
                Al[kt].h[8 + i] = (__bf16)(f1 - (float)h1);
            }
        }
    }

    float minv[8];
    int   mini[8];
    #pragma unroll
    for (int k = 0; k < 8; ++k) { minv[k] = 3.4e38f; mini[k] = 0; }

    // stage chunk `ch` into LDS buffer `buf` (async when available)
    auto stage = [&](int buf, int ch) {
        const uint4* gh = (const uint4*)(cb_hi + (size_t)ch * CHUNK * HIDDEN);
        const uint4* gl = (const uint4*)(cb_lo + (size_t)ch * CHUNK * HIDDEN);
        uint4* sh = (uint4*)lds_hi[buf];
        uint4* sl = (uint4*)lds_lo[buf];
        #pragma unroll
        for (int e = tid; e < CHUNK * HIDDEN / 8; e += 256) {
            async_copy_b128(gh + e, sh + e);
            async_copy_b128(gl + e, sl + e);
        }
    };

    stage(0, 0);
    wait_async_copies();
    __syncthreads();

    for (int chunk = 0; chunk < NCHUNKS; ++chunk) {
        const int buf = chunk & 1;
        if (chunk + 1 < NCHUNKS) stage(buf ^ 1, chunk + 1);   // overlap with compute

        for (int nt = 0; nt < CHUNK / 16; ++nt) {
            // three independent accumulators -> no back-to-back WMMA RAW chain
            v8f acc0 = {0.f, 0.f, 0.f, 0.f, 0.f, 0.f, 0.f, 0.f};
            v8f acc1 = {0.f, 0.f, 0.f, 0.f, 0.f, 0.f, 0.f, 0.f};
            v8f acc2 = {0.f, 0.f, 0.f, 0.f, 0.f, 0.f, 0.f, 0.f};
            const int col   = nt * 16 + (lane & 15);       // code row within chunk
            const int koffB = (lane >= 16) ? 8 : 0;        // B: table order (8h+s, 16+8h+s)
            #pragma unroll
            for (int kt = 0; kt < 8; ++kt) {
                Frag Bh, Bl;
                const __bf16* ph = &lds_hi[buf][col * HIDDEN + kt * 32 + koffB];
                const __bf16* pl = &lds_lo[buf][col * HIDDEN + kt * 32 + koffB];
                Bh.q[0] = *(const uint4*)(ph);     Bh.q[1] = *(const uint4*)(ph + 16);
                Bl.q[0] = *(const uint4*)(pl);     Bl.q[1] = *(const uint4*)(pl + 16);
                acc0 = wmma_bf16(Ah[kt].v, Bh.v, acc0);   // zh.ch
                acc1 = wmma_bf16(Ah[kt].v, Bl.v, acc1);   // zh.cl
                acc2 = wmma_bf16(Al[kt].v, Bh.v, acc2);   // zl.ch
            }
            const int code = chunk * CHUNK + nt * 16 + (lane & 15);
            const float cn = cnorm[code];
            #pragma unroll
            for (int k = 0; k < 8; ++k) {
                float d = cn - 2.0f * (acc0[k] + acc1[k] + acc2[k]);
                if (d < minv[k]) { minv[k] = d; mini[k] = code; }   // strict < keeps lowest idx
            }
        }
        wait_async_copies();   // next buffer fully staged (this wave's copies)
        __syncthreads();       // ... and everyone else's; all reads of current buf done
    }

    // 16-lane min+index reduction within each half-wave (xor<=8 stays in half)
    #pragma unroll
    for (int k = 0; k < 8; ++k) {
        float v = minv[k];
        int   i = mini[k];
        #pragma unroll
        for (int off = 8; off >= 1; off >>= 1) {
            float vo = __shfl_xor(v, off, 32);
            int   io = __shfl_xor(i, off, 32);
            if (vo < v || (vo == v && io < i)) { v = vo; i = io; }
        }
        if (lane == 0)  idx_out[nodeBase + k]     = i;   // M = k (lanes 0-15)
        if (lane == 16) idx_out[nodeBase + 8 + k] = i;   // M = k+8 (lanes 16-31)
    }
}

// ---------------------------------------------------------------------------
// Kernel 3: per-node scatter. One wave per node (8 nodes / block).
//   vq_accum += ||q-zt||^2 + ||q-zs||^2 ;  hist[g,j]++ ;  counts[g]++ ;
//   ce_graph[g] += lse[j] - logits[j, atomic_num-1]
// ---------------------------------------------------------------------------
__global__ __launch_bounds__(256) void scatter_kernel(
    const float* __restrict__ zt, const float* __restrict__ zs,
    const float* __restrict__ cb, const int* __restrict__ idx,
    const int* __restrict__ batch, const int* __restrict__ atomic_nums,
    const float* __restrict__ logits, const float* __restrict__ lse,
    unsigned* __restrict__ hist, unsigned* __restrict__ counts,
    float* __restrict__ ce_graph, float* __restrict__ vq_accum)
{
    const int lane = threadIdx.x & 31;
    const int node = blockIdx.x * 8 + (threadIdx.x >> 5);
    const int j = idx[node];
    const float* q = cb + (size_t)j * HIDDEN;
    const float* a = zt + (size_t)node * HIDDEN;
    const float* b = zs + (size_t)node * HIDDEN;
    float s = 0.f;
    #pragma unroll
    for (int i = lane; i < HIDDEN; i += 32) {
        float qv = q[i];
        float d0 = qv - a[i], d1 = qv - b[i];
        s += d0 * d0 + d1 * d1;
    }
    #pragma unroll
    for (int off = 16; off >= 1; off >>= 1) s += __shfl_xor(s, off, 32);
    if (lane == 0) {
        atomicAdd(vq_accum, s);
        const int g = batch[node];
        atomicAdd(&hist[(size_t)g * N_CODES + j], 1u);
        atomicAdd(&counts[g], 1u);
        float ce = lse[j] - logits[j * MAXZ + (atomic_nums[node] - 1)];
        atomicAdd(&ce_graph[g], ce);
    }
}

// ---------------------------------------------------------------------------
// Kernel 4: z2sum = hist @ codebook  via WMMA  (2048x512 @ 512x256).
// hist counts are small ints -> exact in bf16; codebook uses hi/lo split.
// One wave per 16x16 output tile: 128 m-tiles x 16 n-tiles = 2048 waves.
// ---------------------------------------------------------------------------
__global__ __launch_bounds__(256) void z2mm_kernel(
    const unsigned* __restrict__ hist, const __bf16* __restrict__ cb_hi,
    const __bf16* __restrict__ cb_lo, float* __restrict__ z2sum)
{
    const int tid  = threadIdx.x;
    const int lane = tid & 31;
    const int wg   = blockIdx.x * 8 + (tid >> 5);   // 0..2047
    const int mt   = wg >> 4;                       // graph tile
    const int nt   = wg & 15;                       // hidden tile
    const int g    = mt * 16 + (lane & 15);
    const int h    = nt * 16 + (lane & 15);
    const int koffA = (lane >= 16) ? 8 : 0;
    const int koffB = (lane >= 16) ? 8 : 0;

    v8f accH = {0.f, 0.f, 0.f, 0.f, 0.f, 0.f, 0.f, 0.f};
    v8f accL = {0.f, 0.f, 0.f, 0.f, 0.f, 0.f, 0.f, 0.f};
    for (int kt = 0; kt < N_CODES / 32; ++kt) {
        Frag A, Bh, Bl;
        const unsigned* hrow = hist + (size_t)g * N_CODES + kt * 32 + koffA;
        #pragma unroll
        for (int i = 0; i < 8; ++i) {
            A.h[i]     = (__bf16)(float)hrow[i];
            A.h[8 + i] = (__bf16)(float)hrow[16 + i];
        }
        const int klo = kt * 32 + koffB;
        #pragma unroll
        for (int e = 0; e < 16; ++e) {
            Bh.h[e] = cb_hi[(size_t)(klo + (e < 8 ? e : e + 8)) * HIDDEN + h];
            Bl.h[e] = cb_lo[(size_t)(klo + (e < 8 ? e : e + 8)) * HIDDEN + h];
        }
        accH = wmma_bf16(A.v, Bh.v, accH);
        accL = wmma_bf16(A.v, Bl.v, accL);
    }
    #pragma unroll
    for (int k = 0; k < 8; ++k) {
        const int gg = mt * 16 + k + ((lane >= 16) ? 8 : 0);
        z2sum[(size_t)gg * HIDDEN + h] = accH[k] + accL[k];
    }
}

// ---------------------------------------------------------------------------
// Kernel 5: finalize. Lattice head + mse, atom loss, combine. 1 block.
// ---------------------------------------------------------------------------
__global__ __launch_bounds__(256) void finalize_kernel(
    const float* __restrict__ z2sum, const unsigned* __restrict__ counts,
    const float* __restrict__ ce_graph, const float* __restrict__ vq_accum,
    const float* __restrict__ W_latt, const float* __restrict__ b_latt,
    const float* __restrict__ s_mean, const float* __restrict__ s_std,
    const float* __restrict__ lsc, float* __restrict__ out)
{
    __shared__ float rl[256], ra[256];
    const int t = threadIdx.x;
    float lsum = 0.f, asum = 0.f;
    for (int g = t; g < N_GRAPHS; g += 256) {
        float cnt  = (float)counts[g];
        float safe = cnt > 1.f ? cnt : 1.f;
        float inv  = 1.f / safe;
        const float* zrow = z2sum + (size_t)g * HIDDEN;
        float pred[6];
        #pragma unroll
        for (int i = 0; i < 6; ++i) pred[i] = b_latt[i];
        for (int hh = 0; hh < HIDDEN; ++hh) {
            float z = zrow[hh] * inv;
            #pragma unroll
            for (int i = 0; i < 6; ++i) pred[i] += z * W_latt[hh * 6 + i];
        }
        #pragma unroll
        for (int i = 0; i < 6; ++i) {
            float target = (lsc[g * 6 + i] - s_mean[i]) / s_std[i];
            float d = pred[i] - target;
            lsum += d * d;
        }
        asum += ce_graph[g] * inv;
    }
    rl[t] = lsum; ra[t] = asum;
    __syncthreads();
    for (int s = 128; s > 0; s >>= 1) {
        if (t < s) { rl[t] += rl[t + s]; ra[t] += ra[t + s]; }
        __syncthreads();
    }
    if (t == 0) {
        float vq   = 1.25f * vq_accum[0] / ((float)N_NODES * (float)HIDDEN);
        float latt = rl[0] / ((float)N_GRAPHS * 6.0f);
        float atom = ra[0] / (float)N_GRAPHS;
        out[0] = vq + atom + latt;
    }
}

// ---------------------------------------------------------------------------
extern "C" void kernel_launch(void* const* d_in, const int* in_sizes, int n_in,
                              void* d_out, int out_size, void* d_ws, size_t ws_size,
                              hipStream_t stream)
{
    const float* zt      = (const float*)d_in[0];
    const float* zs      = (const float*)d_in[1];
    const float* cb      = (const float*)d_in[2];
    const float* W_latt  = (const float*)d_in[3];
    const float* b_latt  = (const float*)d_in[4];
    const float* W_atom  = (const float*)d_in[5];
    const float* b_atom  = (const float*)d_in[6];
    const float* s_mean  = (const float*)d_in[7];
    const float* s_std   = (const float*)d_in[8];
    const float* lsc     = (const float*)d_in[9];
    const int*   batch   = (const int*)d_in[10];
    // d_in[11] = num_atoms (only feeds pred_lengths, unused by the loss)
    const int*   at_nums = (const int*)d_in[12];
    float* out = (float*)d_out;

    // workspace layout (all offsets multiples of 256 bytes)
    char* ws = (char*)d_ws;
    size_t o = 0;
    unsigned* hist     = (unsigned*)(ws + o); o += (size_t)N_GRAPHS * N_CODES * 4; // 4 MB
    unsigned* counts   = (unsigned*)(ws + o); o += (size_t)N_GRAPHS * 4;           // 8 KB
    float*    ce_graph = (float*)(ws + o);    o += (size_t)N_GRAPHS * 4;           // 8 KB
    float*    vq_accum = (float*)(ws + o);    o += 256;
    const size_t zero_bytes = o;                                                   // zeroed region
    int*      idx      = (int*)(ws + o);      o += (size_t)N_NODES * 4;            // 512 KB
    __bf16*   cb_hi    = (__bf16*)(ws + o);   o += (size_t)N_CODES * HIDDEN * 2;   // 256 KB
    __bf16*   cb_lo    = (__bf16*)(ws + o);   o += (size_t)N_CODES * HIDDEN * 2;   // 256 KB
    float*    cnorm    = (float*)(ws + o);    o += (size_t)N_CODES * 4;            // 2 KB
    float*    logits   = (float*)(ws + o);    o += (size_t)N_CODES * MAXZ * 4;     // 200 KB
    float*    lse      = (float*)(ws + o);    o += (size_t)N_CODES * 4;            // 2 KB
    float*    z2sum    = (float*)(ws + o);    o += (size_t)N_GRAPHS * HIDDEN * 4;  // 2 MB

    (void)hipMemsetAsync(ws, 0, zero_bytes, stream);

    precompute_codes<<<N_CODES, 256, 0, stream>>>(cb, W_atom, b_atom,
                                                  cb_hi, cb_lo, cnorm, logits, lse);

    argmin_kernel<<<N_NODES / (8 * 16), 256, 0, stream>>>(zt, cb_hi, cb_lo, cnorm, idx);

    scatter_kernel<<<N_NODES / 8, 256, 0, stream>>>(zt, zs, cb, idx, batch, at_nums,
                                                    logits, lse, hist, counts,
                                                    ce_graph, vq_accum);

    z2mm_kernel<<<(N_GRAPHS / 16) * (HIDDEN / 16) / 8, 256, 0, stream>>>(hist, cb_hi,
                                                                         cb_lo, z2sum);

    finalize_kernel<<<1, 256, 0, stream>>>(z2sum, counts, ce_graph, vq_accum,
                                           W_latt, b_latt, s_mean, s_std, lsc, out);
}
